// DeformConvDownSample_17317308137713
// MI455X (gfx1250) — compile-verified, matrix operand order checked by reading, not run on hardware
//
#include <hip/hip_runtime.h>
#include <hip/hip_bf16.h>

// ---------------------------------------------------------------------------
// Types for CDNA5 WMMA (wave32): v_wmma_f32_16x16x32_bf16
// ---------------------------------------------------------------------------
typedef __attribute__((ext_vector_type(16))) __bf16        v16bf;
typedef __attribute__((ext_vector_type(8)))  float         v8f;
typedef __attribute__((ext_vector_type(8)))  unsigned int  v8u;
typedef int v4i __attribute__((vector_size(16)));          // matches builtin param

// Async global->LDS (CDNA5, ASYNCcnt-tracked). Signature (probe-confirmed via
// round-2 diagnostics): (v4i AS1* src, v4i AS3* dst, int offset, int cpol).
#if defined(__has_builtin)
#if __has_builtin(__builtin_amdgcn_global_load_async_to_lds_b128) && \
    __has_builtin(__builtin_amdgcn_s_wait_asynccnt)
#define HAVE_ASYNC_LDS 1
#endif
#endif
#ifndef HAVE_ASYNC_LDS
#define HAVE_ASYNC_LDS 0
#endif

#if HAVE_ASYNC_LDS
#define ASYNC_CP16(dst_lds, src_glb)                                          \
    __builtin_amdgcn_global_load_async_to_lds_b128(                           \
        (__attribute__((address_space(1))) v4i*)(unsigned long long)(const void*)(src_glb), \
        (__attribute__((address_space(3))) v4i*)(unsigned int)(unsigned long long)(const void*)(dst_lds), \
        0, 0)
#else
#define ASYNC_CP16(dst_lds, src_glb)                                          \
    (*(v4i*)(dst_lds) = *(const v4i*)(src_glb))
#endif

__device__ __forceinline__ unsigned short f2bf(float f) {
    unsigned int u = __builtin_bit_cast(unsigned int, f);
    unsigned int r = u + 0x7FFFu + ((u >> 16) & 1u);   // round-to-nearest-even
    return (unsigned short)(r >> 16);
}

__device__ __forceinline__ v8f wmma_bf16(v16bf a, v16bf b, v8f c) {
    return __builtin_amdgcn_wmma_f32_16x16x32_bf16(
        /*neg_a=*/false, a, /*neg_b=*/false, b,
        /*c_mod=*/(short)0, c, /*reuse_a=*/false, /*reuse_b=*/false);
}

// Load a 16-element bf16 fragment from a row-major [rows][ld] ushort matrix.
// Per-lane K pattern for 16-bit operands (ISA 7.12.2):
//   lanes 0-15 : K = kb..kb+7 , kb+16..kb+23  with kb = k0
//   lanes 16-31: K = kb..kb+7 , kb+16..kb+23  with kb = k0+8
__device__ __forceinline__ v16bf load_frag_rm(const unsigned short* P, int ld,
                                              int row, int kb) {
    const unsigned short* q = P + (size_t)row * ld + kb;
    const unsigned int* p0 = reinterpret_cast<const unsigned int*>(q);
    const unsigned int* p1 = reinterpret_cast<const unsigned int*>(q + 16);
    v8u u;
    u[0] = p0[0]; u[1] = p0[1]; u[2] = p0[2]; u[3] = p0[3];
    u[4] = p1[0]; u[5] = p1[1]; u[6] = p1[2]; u[7] = p1[3];
    return __builtin_bit_cast(v16bf, u);
}

// ---------------------------------------------------------------------------
// Problem constants
// ---------------------------------------------------------------------------
#define BATCH   8
#define CIN     256
#define COUT    256
#define HIN     128
#define WIN     128
#define HO      64
#define WO      64
#define NPIX    (BATCH * HO * WO)          // 32768
#define KDEF    (9 * CIN)                  // 2304
#define KSTEPS  (KDEF / 32)                // 72
#define NEG     0.2f

#define DM_PIX     128                     // pixels per k_deform block
#define DM_THREADS 256                     // 8 waves

// Workspace layout (bytes)
#define WS_OFFS   ((size_t)0)                                    // f32 8*18*64*64
#define WS_BOFF   (WS_OFFS  + (size_t)BATCH*18*HO*WO*4)          // [32][2304] bf16
#define WS_BDEF   (WS_BOFF  + (size_t)32*KDEF*2)                 // [256][2304] bf16
#define WS_B1T    (WS_BDEF  + (size_t)COUT*KDEF*2)               // [256][256] bf16
#define WS_A2     (WS_B1T   + (size_t)COUT*CIN*2)                // [32768][256] bf16
#define WS_Z      (WS_A2    + (size_t)NPIX*COUT*2)               // [b*C][4096] f32
#define WS_STATS  (WS_Z     + (size_t)NPIX*COUT*4)               // [8*256][2] f32

// ---------------------------------------------------------------------------
// Weight convert f32 -> bf16, transposed so B-fragments load as dword pairs.
// ---------------------------------------------------------------------------
__global__ void k_wcvt(const float* __restrict__ w_off,
                       const float* __restrict__ w_def,
                       const float* __restrict__ w_1x1,
                       unsigned short* __restrict__ BoffT,
                       unsigned short* __restrict__ BdefT,
                       unsigned short* __restrict__ B1T) {
    const int tid = blockIdx.x * blockDim.x + threadIdx.x;
    const int nt  = gridDim.x * blockDim.x;
    for (int i = tid; i < 32 * KDEF; i += nt) {
        int o = i / KDEF, idx = i - o * KDEF;
        int kk = idx >> 8, c = idx & 255;
        int ky = kk / 3, kx = kk - ky * 3;
        float v = (o < 18) ? w_off[((o * CIN + c) * 3 + ky) * 3 + kx] : 0.0f;
        BoffT[i] = f2bf(v);
    }
    for (int i = tid; i < COUT * KDEF; i += nt) {
        int o = i / KDEF, idx = i - o * KDEF;
        int kk = idx >> 8, c = idx & 255;
        int ky = kk / 3, kx = kk - ky * 3;
        BdefT[i] = f2bf(w_def[((o * CIN + c) * 3 + ky) * 3 + kx]);
    }
    for (int i = tid; i < COUT * CIN; i += nt)
        B1T[i] = f2bf(w_1x1[i]);
}

__global__ void k_stats_init(float* __restrict__ stats) {
    int i = blockIdx.x * blockDim.x + threadIdx.x;
    if (i < BATCH * COUT * 2) stats[i] = 0.0f;
}

// ---------------------------------------------------------------------------
// offsets = conv3x3(x, w_off, stride 2, pad 1). One wave per 16-pixel tile.
// ---------------------------------------------------------------------------
__global__ __launch_bounds__(32) void k_offsets(const float* __restrict__ x,
                                                const unsigned short* __restrict__ BoffT,
                                                float* __restrict__ offs) {
    __shared__ alignas(16) unsigned short Alds[16 * 32];
    const int lane = threadIdx.x;
    const int m0   = blockIdx.x * 16;
    v8f acc0 = {}, acc1 = {};

    for (int ks = 0; ks < KSTEPS; ++ks) {
        const int k0 = ks * 32;
        const int kk = k0 >> 8;
        const int c0 = k0 & 255;
        const int ky = kk / 3, kx = kk - ky * 3;
        __syncthreads();
        const int c = c0 + lane;
        #pragma unroll 4
        for (int i = 0; i < 16; ++i) {
            int pix = m0 + i;
            int b = pix >> 12, rem = pix & 4095;
            int oy = rem >> 6, ox = rem & 63;
            int iy = oy * 2 - 1 + ky, ix = ox * 2 - 1 + kx;
            float v = 0.0f;
            if ((unsigned)iy < (unsigned)HIN && (unsigned)ix < (unsigned)WIN)
                v = x[(((size_t)b * CIN + c) * HIN + iy) * WIN + ix];
            Alds[i * 32 + lane] = f2bf(v);
        }
        __syncthreads();
        const int kb = k0 + ((lane >> 4) << 3);
        v16bf a  = load_frag_rm(Alds, 32, lane & 15, (lane >> 4) << 3);
        v16bf b0 = load_frag_rm(BoffT, KDEF, (lane & 15),      kb);
        v16bf b1 = load_frag_rm(BoffT, KDEF, 16 + (lane & 15), kb);
        acc0 = wmma_bf16(a, b0, acc0);
        acc1 = wmma_bf16(a, b1, acc1);
    }

    const int n0 = lane & 15, hi = lane >> 4;
    #pragma unroll
    for (int r = 0; r < 8; ++r) {
        int pix = m0 + r + (hi << 3);
        int b = pix >> 12, rem = pix & 4095;
        if (n0 < 18)
            offs[((size_t)b * 18 + n0) * 4096 + rem] = acc0[r];
        if (16 + n0 < 18)
            offs[((size_t)b * 18 + 16 + n0) * 4096 + rem] = acc1[r];
    }
}

// ---------------------------------------------------------------------------
// Deformable conv + leaky ReLU -> bf16 activations A2[NPIX][COUT].
// Block: 256 threads (8 waves), 128 pixels x 256 couts.
// Outer loop over the 9 taps (kk): bilinear coords/weights computed ONCE per
// tap; inner loop over 8 channel-steps does pure gather + WMMA.
// Weight slab (32k x 256n, 16 KB) staged into LDS via async global->LDS DMA,
// double-buffered; shared by all 8 waves.
// ---------------------------------------------------------------------------
__global__ __launch_bounds__(DM_THREADS) void k_deform(const float* __restrict__ x,
                                                       const float* __restrict__ offs,
                                                       const unsigned short* __restrict__ BdefT,
                                                       unsigned short* __restrict__ A2) {
    __shared__ alignas(16) unsigned short Blds[2][COUT][32];   // 32 KB, [n][k]
    __shared__ alignas(16) unsigned short Alds[DM_PIX][32];    // 8 KB,  [m][k]
    const int tid  = threadIdx.x;
    const int lane = tid & 31;
    const int wave = tid >> 5;
    const int m0   = blockIdx.x * DM_PIX;
    v8f acc[16] = {};

    // Stage B slab for K-step `ks` into buffer `buf`: thread t copies row
    // n = t (64 B = 4 x b128). 4 ASYNCcnt entries per wave per stage.
    #define STAGE_B(ks, buf)                                                   \
        do {                                                                   \
            const unsigned short* s0 = BdefT + (size_t)tid * KDEF + (ks) * 32; \
            unsigned short* d0 = &Blds[buf][tid][0];                           \
            ASYNC_CP16(d0,      s0);                                           \
            ASYNC_CP16(d0 + 8,  s0 + 8);                                       \
            ASYNC_CP16(d0 + 16, s0 + 16);                                      \
            ASYNC_CP16(d0 + 24, s0 + 24);                                      \
        } while (0)

    // Per-thread pixel for the A-tile build: 2 threads per pixel.
    const int arow = tid >> 1;               // 0..127
    const int acb  = (tid & 1) * 16;         // channel half: 0 or 16
    const int apix = m0 + arow;
    const int ab   = apix >> 12, arem = apix & 4095;
    const int aoy  = arem >> 6,  aox  = arem & 63;

    STAGE_B(0, 0);

    for (int kk = 0; kk < 9; ++kk) {
        const int ky = kk / 3, kx = kk - ky * 3;

        // ---- bilinear setup: depends only on (pixel, kk) ----
        const size_t ob = ((size_t)ab * 18 + 2 * kk) * 4096 + arem;
        const float dy = offs[ob];
        const float dx = offs[ob + 4096];
        const float py = (float)(aoy * 2 - 1 + ky) + dy;
        const float px = (float)(aox * 2 - 1 + kx) + dx;
        const float y0f = floorf(py), x0f = floorf(px);
        const float wy1 = py - y0f, wy0 = 1.0f - wy1;
        const float wx1 = px - x0f, wx0 = 1.0f - wx1;
        const int y0 = (int)y0f, x0i = (int)x0f;
        const int y1 = y0 + 1, x1 = x0i + 1;
        const float vy0 = ((unsigned)y0  < (unsigned)HIN) ? 1.0f : 0.0f;
        const float vy1 = ((unsigned)y1  < (unsigned)HIN) ? 1.0f : 0.0f;
        const float vx0 = ((unsigned)x0i < (unsigned)WIN) ? 1.0f : 0.0f;
        const float vx1 = ((unsigned)x1  < (unsigned)WIN) ? 1.0f : 0.0f;
        const int y0c = min(max(y0, 0), HIN - 1), y1c = min(max(y1, 0), HIN - 1);
        const int x0c = min(max(x0i, 0), WIN - 1), x1c = min(max(x1, 0), WIN - 1);
        const float w00 = wy0 * wx0 * vy0 * vx0;
        const float w01 = wy0 * wx1 * vy0 * vx1;
        const float w10 = wy1 * wx0 * vy1 * vx0;
        const float w11 = wy1 * wx1 * vy1 * vx1;
        const int o00 = y0c * WIN + x0c, o01 = y0c * WIN + x1c;
        const int o10 = y1c * WIN + x0c, o11 = y1c * WIN + x1c;

        for (int cs = 0; cs < 8; ++cs) {         // channel steps of 32
            const int ks = kk * 8 + cs;
            const int c0 = cs * 32;

            __syncthreads();                     // WAR on Alds/Blds
            if (ks + 1 < KSTEPS) STAGE_B(ks + 1, (ks + 1) & 1);

            // gather 16 channels for this thread's pixel
            const float* xc = x + (((size_t)ab * CIN + c0 + acb) << 14);
            #pragma unroll 4
            for (int j = 0; j < 16; ++j) {
                float s = xc[o00] * w00 + xc[o01] * w01
                        + xc[o10] * w10 + xc[o11] * w11;
                Alds[arow][acb + j] = f2bf(s);
                xc += (size_t)HIN * WIN;
            }

#if HAVE_ASYNC_LDS
            if (ks + 1 < KSTEPS) __builtin_amdgcn_s_wait_asynccnt(4);
            else                 __builtin_amdgcn_s_wait_asynccnt(0);
#endif
            __syncthreads();                     // publish Alds + Blds[ks&1]

            const int kb = (lane >> 4) << 3;
            v16bf a = load_frag_rm(&Alds[0][0], 32, wave * 16 + (lane & 15), kb);
            #pragma unroll
            for (int t = 0; t < 16; ++t) {
                v16bf bf = load_frag_rm(&Blds[ks & 1][0][0], 32,
                                        t * 16 + (lane & 15), kb);
                acc[t] = wmma_bf16(a, bf, acc[t]);
            }
        }
    }

    const int n0 = lane & 15, hi = lane >> 4;
    #pragma unroll
    for (int t = 0; t < 16; ++t) {
        const int n = t * 16 + n0;
        #pragma unroll
        for (int r = 0; r < 8; ++r) {
            const int m = m0 + wave * 16 + r + (hi << 3);
            float v = acc[t][r];
            v = (v >= 0.0f) ? v : NEG * v;
            A2[(size_t)m * COUT + n] = f2bf(v);
        }
    }
    #undef STAGE_B
}

// ---------------------------------------------------------------------------
// 1x1 conv GEMM (bf16 WMMA) + per-(b,c) sum/sumsq. z stored [b*C+c][4096].
// ---------------------------------------------------------------------------
__global__ __launch_bounds__(128) void k_1x1(const unsigned short* __restrict__ A2,
                                             const unsigned short* __restrict__ B1T,
                                             float* __restrict__ z,
                                             float* __restrict__ stats) {
    const int tid   = threadIdx.x;
    const int lane  = tid & 31;
    const int wave  = tid >> 5;
    const int m0    = blockIdx.x * 16;
    const int nbase = wave * 64;
    v8f acc[4] = {};

    const int arow = m0 + (lane & 15);
    #pragma unroll
    for (int ks = 0; ks < 8; ++ks) {
        const int kb = ks * 32 + ((lane >> 4) << 3);
        v16bf a = load_frag_rm(A2, CIN, arow, kb);
        #pragma unroll
        for (int t = 0; t < 4; ++t) {
            v16bf bf = load_frag_rm(B1T, CIN, nbase + t * 16 + (lane & 15), kb);
            acc[t] = wmma_bf16(a, bf, acc[t]);
        }
    }

    const int n0 = lane & 15, hi = lane >> 4;
    const int b  = m0 >> 12;
    const int rem0 = (m0 & 4095) + (hi << 3);
    #pragma unroll
    for (int t = 0; t < 4; ++t) {
        const int n = nbase + t * 16 + n0;
        float s = 0.0f, ss = 0.0f;
        #pragma unroll
        for (int r = 0; r < 8; ++r) {
            const float v = acc[t][r];
            z[(size_t)(b * COUT + n) * 4096 + rem0 + r] = v;
            s += v; ss += v * v;
        }
        atomicAdd(&stats[(size_t)(b * COUT + n) * 2 + 0], s);
        atomicAdd(&stats[(size_t)(b * COUT + n) * 2 + 1], ss);
    }
}

// ---------------------------------------------------------------------------
// Instance norm + leaky ReLU -> d_out (B, COUT, HO, WO). Fully coalesced.
// ---------------------------------------------------------------------------
__global__ __launch_bounds__(256) void k_norm(const float* __restrict__ z,
                                              const float* __restrict__ stats,
                                              float* __restrict__ out) {
    const int bc = blockIdx.x;        // b*256 + c
    const float s    = stats[(size_t)bc * 2 + 0];
    const float ss   = stats[(size_t)bc * 2 + 1];
    const float mean = s * (1.0f / 4096.0f);
    const float var  = ss * (1.0f / 4096.0f) - mean * mean;
    const float inv  = rsqrtf(var + 1e-5f);
    for (int i = threadIdx.x; i < 4096; i += 256) {
        float v = (z[(size_t)bc * 4096 + i] - mean) * inv;
        v = (v >= 0.0f) ? v : NEG * v;
        out[(size_t)bc * 4096 + i] = v;
    }
}

// ---------------------------------------------------------------------------
extern "C" void kernel_launch(void* const* d_in, const int* in_sizes, int n_in,
                              void* d_out, int out_size, void* d_ws, size_t ws_size,
                              hipStream_t stream) {
    (void)in_sizes; (void)n_in; (void)out_size; (void)ws_size;
    const float* x     = (const float*)d_in[0];
    const float* w_off = (const float*)d_in[1];
    const float* w_def = (const float*)d_in[2];
    const float* w_1x1 = (const float*)d_in[3];
    float* out = (float*)d_out;

    char* ws = (char*)d_ws;
    float*          offs  = (float*)(ws + WS_OFFS);
    unsigned short* BoffT = (unsigned short*)(ws + WS_BOFF);
    unsigned short* BdefT = (unsigned short*)(ws + WS_BDEF);
    unsigned short* B1T   = (unsigned short*)(ws + WS_B1T);
    unsigned short* A2    = (unsigned short*)(ws + WS_A2);
    float*          z     = (float*)(ws + WS_Z);
    float*          stats = (float*)(ws + WS_STATS);

    k_wcvt      <<<512,            256,        0, stream>>>(w_off, w_def, w_1x1, BoffT, BdefT, B1T);
    k_stats_init<<<16,             256,        0, stream>>>(stats);
    k_offsets   <<<NPIX / 16,      32,         0, stream>>>(x, BoffT, offs);
    k_deform    <<<NPIX / DM_PIX,  DM_THREADS, 0, stream>>>(x, offs, BdefT, A2);
    k_1x1       <<<NPIX / 16,      128,        0, stream>>>(A2, B1T, z, stats);
    k_norm      <<<BATCH * COUT,   256,        0, stream>>>(z, stats, out);
}
